// InteractionGNN_82678120448604
// MI455X (gfx1250) — matrix operand order repaired and checked by reading, not compiled, not run on hardware
//
#include <hip/hip_runtime.h>
#include <stdint.h>
#include <stddef.h>

// ---------------------------------------------------------------------------
// Types
// ---------------------------------------------------------------------------
typedef __attribute__((ext_vector_type(16))) __bf16 v16bf;
typedef __attribute__((ext_vector_type(8)))  float  v8f;
typedef __attribute__((ext_vector_type(4)))  unsigned int v4u;

#define BM   128      // C-tile rows per block
#define BK   32       // K step (one bf16 WMMA K)
#define HDIM 128      // hidden size (== C-tile cols)
#define LSTR 40       // LDS row stride in bf16 elems (32 + 8 pad -> conflict-free)

union FragB16 { v16bf v; v4u q[2]; };

// ---------------------------------------------------------------------------
// GEMM: C[M x 128] = act( A[M x K] @ W[K x 128] + b )  with fused epilogues.
// A is bf16; optionally A = concat(A0[idx[m]] (K<128), A1[m] (K>=128)).
// WT is the weight stored column-major bf16: WT[n][k], n in [0,128).
// ---------------------------------------------------------------------------
__device__ __forceinline__ void stage_tile(
    const __bf16* __restrict__ A0, const int* __restrict__ idx,
    const __bf16* __restrict__ A1, int a0_stride,
    const __bf16* __restrict__ WT, int wt_stride,
    int M, int blockM, int kt,
    __bf16* sA, __bf16* sB, int tid)
{
  const int k0  = kt * BK;
  const int r   = tid >> 1;   // 0..127 : row of A tile / row of WT
  const int seg = tid & 1;    // each thread moves 32 bytes (16 bf16)

  // ---- A tile (BM x BK), optionally gathered / concatenated ----
  v4u d0 = {0u, 0u, 0u, 0u};
  v4u d1 = {0u, 0u, 0u, 0u};
  const int m = blockM * BM + r;
  if (m < M) {
    const __bf16* p;
    if (A1 && k0 >= HDIM) {
      p = A1 + (size_t)m * HDIM + (k0 - HDIM);
    } else {
      const int row = idx ? idx[m] : m;
      p = A0 + (size_t)row * a0_stride + k0;
    }
    const v4u* g = (const v4u*)(p + seg * 16);
    d0 = g[0]; d1 = g[1];
  }
  v4u* da = (v4u*)(sA + r * LSTR + seg * 16);
  da[0] = d0; da[1] = d1;

  // ---- B tile: rows n=0..127, 32 contiguous K values from WT[n][k0..] ----
  const v4u* gw = (const v4u*)(WT + (size_t)r * wt_stride + k0 + seg * 16);
  v4u w0 = gw[0], w1 = gw[1];
  v4u* db = (v4u*)(sB + r * LSTR + seg * 16);
  db[0] = w0; db[1] = w1;
}

__device__ __forceinline__ void mma_step(const __bf16* sA, const __bf16* sB,
                                         v8f acc[8], int waverow, int lane)
{
  const int hf = lane >> 4;    // half-wave select
  const int ln = lane & 15;

  // A fragment: row (waverow+ln); lanes 0-15 hold K {0..7,16..23},
  // lanes 16-31 hold K {8..15,24..31}  (ISA 16-bit A 16x32 layout).
  FragB16 a;
  const __bf16* ar = sA + (waverow + ln) * LSTR;
  a.q[0] = *(const v4u*)(ar + 8 * hf);
  a.q[1] = *(const v4u*)(ar + 16 + 8 * hf);

#pragma unroll
  for (int f = 0; f < 8; ++f) {
    // B fragment: lane holds column n=f*16+ln, K {8hf..8hf+7, 16+8hf..16+8hf+7}
    FragB16 b;
    const __bf16* br = sB + (f * 16 + ln) * LSTR + 8 * hf;
    b.q[0] = *(const v4u*)(br);
    b.q[1] = *(const v4u*)(br + 16);
    acc[f] = __builtin_amdgcn_wmma_f32_16x16x32_bf16(
        false, a.v, false, b.v, (short)0, acc[f], false, false);
  }
}

__global__ __launch_bounds__(256) void k_gemm_bf16(
    const __bf16* __restrict__ A0, const int* __restrict__ idx,
    const __bf16* __restrict__ A1, int a0_stride,
    const __bf16* __restrict__ WT, int wt_stride, int nk,
    const float* __restrict__ bias, int relu, int M,
    __bf16* __restrict__ outb, float* __restrict__ outf,
    const float* __restrict__ resf,
    __bf16* __restrict__ rmwb,
    float* __restrict__ scatf, const int* __restrict__ scidx)
{
  __shared__ __align__(16) __bf16 sA[2][BM * LSTR];
  __shared__ __align__(16) __bf16 sB[2][BM * LSTR];

  const int tid    = threadIdx.x;
  const int blockM = blockIdx.x;
  const int wave   = tid >> 5;
  const int lane   = tid & 31;
  const int waverow = wave * 16;

  v8f acc[8];
#pragma unroll
  for (int f = 0; f < 8; ++f)
#pragma unroll
    for (int j = 0; j < 8; ++j) acc[f][j] = 0.f;

  stage_tile(A0, idx, A1, a0_stride, WT, wt_stride, M, blockM, 0,
             sA[0], sB[0], tid);
  __syncthreads();

  for (int kt = 0; kt < nk; ++kt) {
    if (kt + 1 < nk)
      stage_tile(A0, idx, A1, a0_stride, WT, wt_stride, M, blockM, kt + 1,
                 sA[(kt + 1) & 1], sB[(kt + 1) & 1], tid);
    mma_step(sA[kt & 1], sB[kt & 1], acc, waverow, lane);
    __syncthreads();
  }

  // ---- epilogue: C element (m = blockM*128 + waverow + 8*hf + j, n = 16f+ln)
  const int hf = lane >> 4;
  const int ln = lane & 15;
#pragma unroll
  for (int f = 0; f < 8; ++f) {
    const int n = f * 16 + ln;
    const float bv = bias[n];
#pragma unroll
    for (int j = 0; j < 8; ++j) {
      const int m = blockM * BM + waverow + hf * 8 + j;
      if (m >= M) continue;
      float v = acc[f][j] + bv;
      if (relu) v = fmaxf(v, 0.f);
      const size_t o = (size_t)m * HDIM + n;
      if (resf) v += resf[o];
      if (rmwb) rmwb[o] = (__bf16)((float)rmwb[o] + v);  // ef += m
      if (scatf) atomicAdd(scatf + (size_t)scidx[m] * HDIM + n, v);
      if (outf) outf[o] = v;
      if (outb) outb[o] = (__bf16)v;
    }
  }
}

// ---------------------------------------------------------------------------
// Helper kernels
// ---------------------------------------------------------------------------
// f32 weight [Ksrc][128] (row-major, (in,out)) -> bf16 WT [128][Kpad] (col-major)
__global__ __launch_bounds__(256) void k_wconv(const float* __restrict__ src,
                                               __bf16* __restrict__ dst,
                                               int Ksrc, int Kpad)
{
  const int t = blockIdx.x * 256 + threadIdx.x;
  if (t >= 128 * Kpad) return;
  const int n = t / Kpad;
  const int k = t - n * Kpad;
  const float v = (k < Ksrc) ? src[(size_t)k * 128 + n] : 0.f;
  dst[t] = (__bf16)v;
}

// f32 [M][Kin] -> bf16 [M][32] zero-padded
__global__ __launch_bounds__(256) void k_padin(const float* __restrict__ src,
                                               __bf16* __restrict__ dst,
                                               size_t total, int Kin)
{
  for (size_t i = (size_t)blockIdx.x * blockDim.x + threadIdx.x; i < total;
       i += (size_t)gridDim.x * blockDim.x) {
    const size_t m = i >> 5;
    const int k = (int)(i & 31);
    dst[i] = (k < Kin) ? (__bf16)src[m * (size_t)Kin + k] : (__bf16)0.f;
  }
}

__global__ __launch_bounds__(256) void k_zero_f32(float* __restrict__ p, size_t n)
{
  for (size_t i = (size_t)blockIdx.x * blockDim.x + threadIdx.x; i < n;
       i += (size_t)gridDim.x * blockDim.x)
    p[i] = 0.f;
}

__global__ __launch_bounds__(256) void k_f2b(const float* __restrict__ s,
                                             __bf16* __restrict__ d, size_t n)
{
  for (size_t i = (size_t)blockIdx.x * blockDim.x + threadIdx.x; i < n;
       i += (size_t)gridDim.x * blockDim.x)
    d[i] = (__bf16)s[i];
}

// Final projection: out[m][0..1] = h[m] @ W[128x2] + b
__global__ __launch_bounds__(256) void k_nodeout2(const __bf16* __restrict__ h,
                                                  const float* __restrict__ Wp,
                                                  const float* __restrict__ bp,
                                                  float* __restrict__ out, int Nn)
{
  __shared__ float sW[256];
  sW[threadIdx.x] = Wp[threadIdx.x];
  __syncthreads();
  const int m = blockIdx.x * 256 + threadIdx.x;
  if (m >= Nn) return;
  float s0 = bp[0], s1 = bp[1];
  const __bf16* hr = h + (size_t)m * HDIM;
#pragma unroll 4
  for (int k = 0; k < HDIM; ++k) {
    const float xv = (float)hr[k];
    s0 += xv * sW[2 * k];
    s1 += xv * sW[2 * k + 1];
  }
  out[2 * m]     = s0;
  out[2 * m + 1] = s1;
}

// ---------------------------------------------------------------------------
// Host launch
// ---------------------------------------------------------------------------
extern "C" void kernel_launch(void* const* d_in, const int* in_sizes, int n_in,
                              void* d_out, int out_size, void* d_ws, size_t ws_size,
                              hipStream_t stream)
{
  (void)n_in; (void)out_size; (void)ws_size;
  const int N = in_sizes[0] / 10;   // NODE_FEAT*WIN = 10
  const int E = in_sizes[1] / 5;    // WIN = 5

  const float* x   = (const float*)d_in[0];
  const float* ea  = (const float*)d_in[1];
  const int*   ei  = (const int*)d_in[2];
  const int*   tgt = ei + E;        // edge_index[1]

  // Detect param flattening order: sorted dict keys (JAX pytree) vs insertion.
  int ei_base, ly_base, ni_base, no_base;
  if (in_sizes[3] == 5 * 128) {     // edge_in first  (sorted keys)
    ei_base = 3; ly_base = 9; ni_base = 69; no_base = 75;
  } else {                          // node_in first  (insertion order)
    ni_base = 3; ei_base = 9; no_base = 15; ly_base = 19;
  }
  auto Wp = [&](int base, int l) { return (const float*)d_in[base + 2 * l]; };
  auto Bp = [&](int base, int l) { return (const float*)d_in[base + 2 * l + 1]; };

  // ---- workspace arena (all offsets 256B aligned) ----
  char* ws = (char*)d_ws;
  size_t off = 0;
  auto alloc = [&](size_t bytes) {
    size_t o = off;
    off += (bytes + 255) & ~(size_t)255;
    return o;
  };
  float*  NF32 = (float*)(ws + alloc((size_t)N * HDIM * 4));
  __bf16* NFB  = (__bf16*)(ws + alloc((size_t)N * HDIM * 2));
  __bf16* EFB  = (__bf16*)(ws + alloc((size_t)E * HDIM * 2));
  float*  AG32 = (float*)(ws + alloc((size_t)N * HDIM * 4));
  __bf16* AGB  = (__bf16*)(ws + alloc((size_t)N * HDIM * 2));
  __bf16* T1   = (__bf16*)(ws + alloc((size_t)E * HDIM * 2));
  __bf16* T2   = (__bf16*)(ws + alloc((size_t)E * HDIM * 2));
  __bf16* XPAD = (__bf16*)(ws + alloc((size_t)N * 32 * 2));
  __bf16* EPAD = (__bf16*)(ws + alloc((size_t)E * 32 * 2));

  // ---- convert weights: f32 [K][128] -> bf16 col-major [128][Kpad] ----
  auto addw = [&](const float* src, int Ksrc, int Kpad) {
    __bf16* d = (__bf16*)(ws + alloc((size_t)128 * Kpad * 2));
    k_wconv<<<(128 * Kpad + 255) / 256, 256, 0, stream>>>(src, d, Ksrc, Kpad);
    return d;
  };
  __bf16* ni[3] = { addw(Wp(ni_base, 0), 10, 32),
                    addw(Wp(ni_base, 1), 128, 128),
                    addw(Wp(ni_base, 2), 128, 128) };
  __bf16* eiw[3] = { addw(Wp(ei_base, 0), 5, 32),
                     addw(Wp(ei_base, 1), 128, 128),
                     addw(Wp(ei_base, 2), 128, 128) };
  __bf16* le[5][3];
  __bf16* ln[5][3];
  for (int i = 0; i < 5; ++i) {
    const int eb = ly_base + 12 * i;      // lin_edge leaves
    const int nb = ly_base + 12 * i + 6;  // lin_node leaves
    for (int l = 0; l < 3; ++l) {
      const int K = (l == 0) ? 256 : 128;
      le[i][l] = addw(Wp(eb, l), K, K);
      ln[i][l] = addw(Wp(nb, l), K, K);
    }
  }
  __bf16* no0 = addw(Wp(no_base, 0), 128, 128);

  // ---- pad/convert inputs to bf16 K=32 ----
  k_padin<<<1024, 256, 0, stream>>>(x,  XPAD, (size_t)N * 32, 10);
  k_padin<<<2048, 256, 0, stream>>>(ea, EPAD, (size_t)E * 32, 5);

  auto gemm = [&](int M, int nk, const __bf16* A0, const int* idx,
                  const __bf16* A1, int a0s, const __bf16* WT, int wts,
                  const float* bias, int relu, __bf16* outb, float* outf,
                  const float* resf, __bf16* rmwb, float* scatf,
                  const int* scidx) {
    const int grid = (M + BM - 1) / BM;
    k_gemm_bf16<<<grid, 256, 0, stream>>>(A0, idx, A1, a0s, WT, wts, nk, bias,
                                          relu, M, outb, outf, resf, rmwb,
                                          scatf, scidx);
  };

  // ---- node encoder: nf = MLP([10->128->128->128])(x) ----
  gemm(N, 1, XPAD, 0, 0, 32,  ni[0], 32,  Bp(ni_base, 0), 1, T1, 0, 0, 0, 0, 0);
  gemm(N, 4, T1,   0, 0, 128, ni[1], 128, Bp(ni_base, 1), 1, T2, 0, 0, 0, 0, 0);
  gemm(N, 4, T2,   0, 0, 128, ni[2], 128, Bp(ni_base, 2), 0, NFB, NF32, 0, 0, 0, 0);

  // ---- edge encoder: ef = MLP([5->128->128->128])(edge_attr) ----
  gemm(E, 1, EPAD, 0, 0, 32,  eiw[0], 32,  Bp(ei_base, 0), 1, T1, 0, 0, 0, 0, 0);
  gemm(E, 4, T1,   0, 0, 128, eiw[1], 128, Bp(ei_base, 1), 1, T2, 0, 0, 0, 0, 0);
  gemm(E, 4, T2,   0, 0, 128, eiw[2], 128, Bp(ei_base, 2), 0, EFB, 0, 0, 0, 0, 0);

  // ---- 5 message-passing layers ----
  for (int i = 0; i < 5; ++i) {
    const int eb = ly_base + 12 * i;
    const int nb = ly_base + 12 * i + 6;
    k_zero_f32<<<2048, 256, 0, stream>>>(AG32, (size_t)N * HDIM);
    // m = edgeMLP(concat(nf[tgt], ef)); fused gather-concat in A loader
    gemm(E, 8, NFB, tgt, EFB, 128, le[i][0], 256, Bp(eb, 0), 1, T1, 0, 0, 0, 0, 0);
    gemm(E, 4, T1,  0,   0,   128, le[i][1], 128, Bp(eb, 1), 1, T2, 0, 0, 0, 0, 0);
    // last edge linear: fused scatter-add(aggr) + ef += m (no m materialized)
    gemm(E, 4, T2,  0,   0,   128, le[i][2], 128, Bp(eb, 2), 0,
         0, 0, 0, EFB, AG32, tgt);
    k_f2b<<<2048, 256, 0, stream>>>(AG32, AGB, (size_t)N * HDIM);
    // nf = nf + nodeMLP(concat(nf, aggr))
    gemm(N, 8, NFB, 0, AGB, 128, ln[i][0], 256, Bp(nb, 0), 1, T1, 0, 0, 0, 0, 0);
    gemm(N, 4, T1,  0, 0,   128, ln[i][1], 128, Bp(nb, 1), 1, T2, 0, 0, 0, 0, 0);
    gemm(N, 4, T2,  0, 0,   128, ln[i][2], 128, Bp(nb, 2), 0,
         NFB, NF32, NF32, 0, 0, 0);
  }

  // ---- decoder: out = Linear(2) ( relu(Linear(128)(nf)) ) ----
  gemm(N, 4, NFB, 0, 0, 128, no0, 128, Bp(no_base, 0), 1, T1, 0, 0, 0, 0, 0);
  k_nodeout2<<<(N + 255) / 256, 256, 0, stream>>>(
      T1, Wp(no_base, 1), Bp(no_base, 1), (float*)d_out, N);
}
